// MultiHeadAttention2_59081570124597
// MI455X (gfx1250) — matrix operand order repaired and checked by reading, not compiled, not run on hardware
//
#include <hip/hip_runtime.h>
#include <hip/hip_bf16.h>

typedef __attribute__((ext_vector_type(16))) __bf16 v16bf;
typedef __attribute__((ext_vector_type(8)))  __bf16 v8bf;
typedef __attribute__((ext_vector_type(4)))  __bf16 v4bf;
typedef __attribute__((ext_vector_type(8)))  float  v8f;

// D = A(16x32 bf16) * B(32x16 bf16) + C(16x16 f32)
__device__ __forceinline__ v8f wmma_bf16(v16bf a, v16bf b, v8f c) {
    return __builtin_amdgcn_wmma_f32_16x16x32_bf16(
        /*neg_a=*/false, a, /*neg_b=*/false, b,
        /*c_mod=*/(short)0, c, /*reuse_a=*/false, /*reuse_b=*/false);
}

// A-fragment from a row-major 16x32 bf16 tile (row stride 32 elements).
// lane m = lane&15 holds row m; element e -> K = (lane>>4)*16 + e
__device__ __forceinline__ v16bf load_a_rm32(const __bf16* t, int lane) {
    int m  = lane & 15;
    int hf = lane >> 4;
    const __bf16* p = t + m * 32 + hf * 16;
    return *(const v16bf*)(p);
}

// B-fragment from a "B^T row-major" 16x32 tile: row n = column n of B,
// 32 contiguous contraction values; lane n&15 takes half hf*16.
__device__ __forceinline__ v16bf load_b_tr32(const __bf16* t, int lane) {
    int n  = lane & 15;
    int hf = lane >> 4;
    return *(const v16bf*)(t + n * 32 + hf * 16);
}

// Async copy 16 bytes global -> LDS (no VGPR staging, tracked by ASYNCcnt).
__device__ __forceinline__ void async_b128(uint32_t lds_off, const void* gptr) {
    asm volatile("global_load_async_to_lds_b128 %0, %1, off"
                 :: "v"(lds_off), "v"(gptr)
                 : "memory");
}
__device__ __forceinline__ uint32_t lds_off32(const void* p) {
    return (uint32_t)(uintptr_t)p;   // LDS offset lives in the low 32 bits
}

// ---------------- f32 -> bf16 convert (4 elems/thread) ----------------
__global__ void cvt_f32_bf16(const float* __restrict__ in,
                             __bf16* __restrict__ out, int n4) {
    int i = blockIdx.x * blockDim.x + threadIdx.x;
    if (i < n4) {
        float4 v = ((const float4*)in)[i];
        v4bf o = { (__bf16)v.x, (__bf16)v.y, (__bf16)v.z, (__bf16)v.w };
        ((v4bf*)out)[i] = o;
    }
}

// ---------------- projection GEMM: Y = X @ W^T ----------------
// X: [4096,1024] bf16 row-major, W: [1024,1024] bf16 row-major ([out,in]).
// Block tile 64(M) x 128(N), 4 waves, wave tile 32x64 (8 WMMAs / k-step).
// Double-buffered LDS fed by GLOBAL_LOAD_ASYNC_TO_LDS_B128 (ASYNCcnt).
// MODE 0: Y bf16 [4096,1024] row-major. MODE 1 (V): Y = vT[b][h][u][t].
template<int MODE>
__global__ __launch_bounds__(128)
void gemm_proj(const __bf16* __restrict__ X, const __bf16* __restrict__ W,
               __bf16* __restrict__ Y) {
    __shared__ __bf16 As[2][64 * 32];    // 2 x 4 KB
    __shared__ __bf16 Bs[2][128 * 32];   // 2 x 8 KB
    const int tid  = threadIdx.x;
    const int lane = tid & 31;
    const int wave = tid >> 5;
    const int m0 = blockIdx.x * 64;
    const int n0 = blockIdx.y * 128;
    const int waveM = (wave >> 1) * 32;
    const int waveN = (wave & 1) * 64;

    v8f acc[2][4];
#pragma unroll
    for (int i = 0; i < 2; ++i)
#pragma unroll
        for (int j = 0; j < 4; ++j) acc[i][j] = (v8f){};

    // cooperative-load mapping: A row ar (2 lanes/row), B row tid
    const int ar = tid >> 1;
    const int ah = tid & 1;
    const __bf16* gA = &X[(size_t)(m0 + ar) * 1024 + ah * 16];
    const __bf16* gB = &W[(size_t)(n0 + tid) * 1024];

    const uint32_t ldsA = lds_off32(&As[0][ar * 32 + ah * 16]);
    const uint32_t ldsB = lds_off32(&Bs[0][tid * 32]);
    const uint32_t bufA = lds_off32(&As[1][0]) - lds_off32(&As[0][0]);
    const uint32_t bufB = lds_off32(&Bs[1][0]) - lds_off32(&Bs[0][0]);

    // prologue: stage k0 = 0 into buffer 0
    async_b128(ldsA,      gA);
    async_b128(ldsA + 16, gA + 8);        // +8 bf16 = 16 B
    async_b128(ldsB,      gB);
    async_b128(ldsB + 16, gB + 8);
    async_b128(ldsB + 32, gB + 16);
    async_b128(ldsB + 48, gB + 24);

    uint32_t cur = 0;
    for (int k0 = 0; k0 < 1024; k0 += 32) {
        asm volatile("s_wait_asynccnt 0x0" ::: "memory");
        __syncthreads();

        if (k0 + 32 < 1024) {   // stage next chunk into the other buffer
            uint32_t oa = ldsA + (cur ^ 1) * bufA;
            uint32_t ob = ldsB + (cur ^ 1) * bufB;
            const __bf16* ga = gA + k0 + 32;
            const __bf16* gb = gB + k0 + 32;
            async_b128(oa,      ga);
            async_b128(oa + 16, ga + 8);
            async_b128(ob,      gb);
            async_b128(ob + 16, gb + 8);
            async_b128(ob + 32, gb + 16);
            async_b128(ob + 48, gb + 24);
        }

        v16bf a0 = load_a_rm32(&As[cur][(waveM +  0) * 32], lane);
        v16bf a1 = load_a_rm32(&As[cur][(waveM + 16) * 32], lane);
        v16bf bf[4];
#pragma unroll
        for (int j = 0; j < 4; ++j)
            bf[j] = load_b_tr32(&Bs[cur][(waveN + j * 16) * 32], lane);

#pragma unroll
        for (int j = 0; j < 4; ++j) {
            acc[0][j] = wmma_bf16(a0, bf[j], acc[0][j]);
            acc[1][j] = wmma_bf16(a1, bf[j], acc[1][j]);
        }
        cur ^= 1;
    }

    const int cn = lane & 15;
    const int hf = lane >> 4;
#pragma unroll
    for (int ti = 0; ti < 2; ++ti)
#pragma unroll
        for (int tj = 0; tj < 4; ++tj) {
#pragma unroll
            for (int r = 0; r < 8; ++r) {
                int row = m0 + waveM + ti * 16 + r + 8 * hf;   // token index
                int col = n0 + waveN + tj * 16 + cn;           // unit index
                __bf16 v = (__bf16)acc[ti][tj][r];
                if (MODE == 0) {
                    Y[(size_t)row * 1024 + col] = v;
                } else {
                    int b = row >> 11, t = row & 2047;
                    int h = col >> 7,  u = col & 127;
                    Y[(((size_t)(b * 8 + h) * 128 + u) << 11) + t] = v;
                }
            }
        }
}

// ---------------- flash attention ----------------
// Qb/Kb: bf16 [2*2048, 1024] row-major. Vt: bf16 [2][8][128][2048].
// Out: f32 [2, 2048, 1024]. Grid (32, 8, 2), 128 threads (4 waves x 16 q-rows).
__global__ __launch_bounds__(128)
void attn_flash(const __bf16* __restrict__ Qb, const __bf16* __restrict__ Kb,
                const __bf16* __restrict__ Vt, float* __restrict__ Out) {
    __shared__ __bf16 Plds[4][16 * 32];
    const int tid  = threadIdx.x;
    const int lane = tid & 31;
    const int wave = tid >> 5;
    const int h = blockIdx.y;
    const int b = blockIdx.z;
    const int q0 = blockIdx.x * 64 + wave * 16;

    const int ln = lane & 15;
    const int hf = lane >> 4;

    // Q tile: 16 rows x 128 head-dims as 4 A-fragments, kept in registers
    v16bf qa[4];
    {
        const __bf16* qbase = Qb + (size_t)(b * 2048 + q0 + ln) * 1024 + h * 128;
#pragma unroll
        for (int c = 0; c < 4; ++c) {
            const __bf16* p = qbase + c * 32 + hf * 16;
            qa[c] = *(const v16bf*)(p);
        }
    }

    v8f o[8];
#pragma unroll
    for (int j = 0; j < 8; ++j) o[j] = (v8f){};
    float mrun[8], lrun[8];
#pragma unroll
    for (int r = 0; r < 8; ++r) { mrun[r] = -3.0e38f; lrun[r] = 0.0f; }

    const __bf16* kbase = Kb + (size_t)(b * 2048) * 1024 + h * 128;
    const __bf16* vbase = Vt + (size_t)(b * 8 + h) * 128 * 2048;
    __bf16* pl = &Plds[wave][0];

    for (int t0 = 0; t0 < 2048; t0 += 32) {
        if (t0 + 32 < 2048) {
            __builtin_prefetch(kbase + (size_t)(t0 + 32 + lane) * 1024, 0, 3);
            __builtin_prefetch(vbase + (size_t)(lane * 4) * 2048 + t0 + 32, 0, 3);
        }

        // ---- batch-load all 8 K B-fragments, then 8 score WMMAs ----
        v16bf kb0[4], kb1[4];
#pragma unroll
        for (int c = 0; c < 4; ++c) {
            kb0[c] = *(const v16bf*)(kbase + (size_t)(t0 + ln) * 1024 + c * 32 + hf * 16);
            kb1[c] = *(const v16bf*)(kbase + (size_t)(t0 + 16 + ln) * 1024 + c * 32 + hf * 16);
        }
        v8f s0 = {}, s1 = {};
#pragma unroll
        for (int c = 0; c < 4; ++c) {
            s0 = wmma_bf16(qa[c], kb0[c], s0);
            s1 = wmma_bf16(qa[c], kb1[c], s1);
        }

        // ---- issue V B-fragment loads now; they drain behind the softmax ----
        v16bf bv[8];
#pragma unroll
        for (int j = 0; j < 8; ++j)
            bv[j] = *(const v16bf*)(vbase + (size_t)(j * 16 + ln) * 2048 + t0 + hf * 16);

        // online softmax (scale = 1/sqrt(1024) = 1/32); rows live in 16-lane halves
        float alpha[8], p0[8], p1[8];
#pragma unroll
        for (int r = 0; r < 8; ++r) {
            float x0 = s0[r] * 0.03125f;
            float x1 = s1[r] * 0.03125f;
            float mx = fmaxf(x0, x1);
            mx = fmaxf(mx, __shfl_xor(mx, 1, 32));
            mx = fmaxf(mx, __shfl_xor(mx, 2, 32));
            mx = fmaxf(mx, __shfl_xor(mx, 4, 32));
            mx = fmaxf(mx, __shfl_xor(mx, 8, 32));
            float mnew = fmaxf(mrun[r], mx);
            alpha[r] = __expf(mrun[r] - mnew);
            p0[r] = __expf(x0 - mnew);
            p1[r] = __expf(x1 - mnew);
            float rs = p0[r] + p1[r];
            rs += __shfl_xor(rs, 1, 32);
            rs += __shfl_xor(rs, 2, 32);
            rs += __shfl_xor(rs, 4, 32);
            rs += __shfl_xor(rs, 8, 32);
            lrun[r] = lrun[r] * alpha[r] + rs;
            mrun[r] = mnew;
        }
#pragma unroll
        for (int j = 0; j < 8; ++j)
#pragma unroll
            for (int r = 0; r < 8; ++r) o[j][r] *= alpha[r];

        // transpose P (D-layout -> A-layout) through per-wave LDS
#pragma unroll
        for (int r = 0; r < 8; ++r) {
            pl[(r + 8 * hf) * 32 + ln]      = (__bf16)p0[r];
            pl[(r + 8 * hf) * 32 + 16 + ln] = (__bf16)p1[r];
        }
        asm volatile("s_wait_dscnt 0" ::: "memory");
        v16bf pa = load_a_rm32(pl, lane);

        // O += P @ V over 8 feature tiles
#pragma unroll
        for (int j = 0; j < 8; ++j)
            o[j] = wmma_bf16(pa, bv[j], o[j]);
    }

    float inv[8];
#pragma unroll
    for (int r = 0; r < 8; ++r) inv[r] = 1.0f / lrun[r];
#pragma unroll
    for (int j = 0; j < 8; ++j)
#pragma unroll
        for (int r = 0; r < 8; ++r) {
            int q = q0 + r + 8 * hf;
            int u = h * 128 + j * 16 + ln;
            Out[(size_t)(b * 2048 + q) * 1024 + u] = o[j][r] * inv[r];
        }
}

extern "C" void kernel_launch(void* const* d_in, const int* in_sizes, int n_in,
                              void* d_out, int out_size, void* d_ws, size_t ws_size,
                              hipStream_t stream) {
    const float* query = (const float*)d_in[0];   // [2,2048,1024]
    const float* keys  = (const float*)d_in[1];   // [2,2048,1024]
    const float* Wq    = (const float*)d_in[2];   // [1024,1024]
    const float* Wk    = (const float*)d_in[3];
    const float* Wv    = (const float*)d_in[4];
    float* out = (float*)d_out;

    char* ws = (char*)d_ws;
    __bf16* Xq  = (__bf16*)(ws + 0);          // 8 MB  query bf16
    __bf16* Xk  = (__bf16*)(ws + 8388608);    // 8 MB  keys  bf16
    __bf16* Wqb = (__bf16*)(ws + 16777216);   // 2 MB
    __bf16* Wkb = (__bf16*)(ws + 18874368);   // 2 MB
    __bf16* Wvb = (__bf16*)(ws + 20971520);   // 2 MB
    __bf16* Qb  = (__bf16*)(ws + 23068672);   // 8 MB  Q proj bf16
    __bf16* Kb  = (__bf16*)(ws + 31457280);   // 8 MB  K proj bf16
    __bf16* Vt  = (__bf16*)(ws + 39845888);   // 8 MB  V proj, feature-major

    {
        int n4 = (2 * 2048 * 1024) / 4;
        cvt_f32_bf16<<<(n4 + 255) / 256, 256, 0, stream>>>(query, Xq, n4);
        cvt_f32_bf16<<<(n4 + 255) / 256, 256, 0, stream>>>(keys,  Xk, n4);
        int w4 = (1024 * 1024) / 4;
        cvt_f32_bf16<<<(w4 + 255) / 256, 256, 0, stream>>>(Wq, Wqb, w4);
        cvt_f32_bf16<<<(w4 + 255) / 256, 256, 0, stream>>>(Wk, Wkb, w4);
        cvt_f32_bf16<<<(w4 + 255) / 256, 256, 0, stream>>>(Wv, Wvb, w4);
    }

    dim3 gg(64, 8);  // 4096/64 x 1024/128
    gemm_proj<0><<<gg, 128, 0, stream>>>(Xq, Wqb, Qb);
    gemm_proj<0><<<gg, 128, 0, stream>>>(Xk, Wkb, Kb);
    gemm_proj<1><<<gg, 128, 0, stream>>>(Xk, Wvb, Vt);

    dim3 ga(32, 8, 2);  // q-blocks x heads x batch
    attn_flash<<<ga, 128, 0, stream>>>(Qb, Kb, Vt, out);
}
